// MultiHeadAttentionLayer_53326313947334
// MI455X (gfx1250) — matrix operand order repaired and checked by reading, not compiled, not run on hardware
//
#include <hip/hip_runtime.h>
#include <hip/hip_bf16.h>

// ---------------------------------------------------------------------------
// Graph multi-head attention layer for MI455X (gfx1250, wave32, WMMA).
//   Qh/Kh/Vh = node_feats @ W{q,k,v} + b      (WMMA f16 -> f32)
//   proj_e   = edge_feats @ We + be           (WMMA f16 -> f32, streamed via
//                                              GLOBAL_LOAD_ASYNC_TO_LDS dbuf)
//   score    = clip(K[src]*Q[dst]/sqrt(D)) * proj_e   -> e_out
//   w        = exp(clip(sum_D score))
//   h_out    = segsum(w*V[src], dst) / (segsum(w, dst) + 1e-8)
// ---------------------------------------------------------------------------

#define N_NODES 50000
#define N_EDGES 800000
#define F_IN    128
#define HD      128       // H*D = 8*16
#define ET_PER_BLK 8      // edge tiles (of 16 edges) per block

typedef _Float16 v16h __attribute__((ext_vector_type(16)));
typedef _Float16 h8   __attribute__((ext_vector_type(8)));
typedef float    v8f  __attribute__((ext_vector_type(8)));
typedef float    f4   __attribute__((ext_vector_type(4)));

__device__ __forceinline__ unsigned lds_off32(const void* p) {
  // Generic pointers to LDS carry the LDS byte offset in the low 32 bits
  // (aperture maps ADDR[31:0] -> LDS_ADDR, CDNA5 ISA 10.2).
  return (unsigned)(unsigned long long)p;
}

// ---------------------------------------------------------------------------
// Prep: pack Wq|Wk|Wv|We into f16, output-column-major: Wt[c][k], c in [0,512)
// so each WMMA B-fragment (lane n, K 0..15 / 16..31) is one contiguous 32B.
// ---------------------------------------------------------------------------
__global__ void __launch_bounds__(128) prep_weights_kernel(
    const float* __restrict__ Wq, const float* __restrict__ Wk,
    const float* __restrict__ Wv, const float* __restrict__ We,
    _Float16* __restrict__ Wt) {
  int c = blockIdx.x;          // 0..511 output column
  int k = threadIdx.x;         // 0..127 reduction index
  const float* W = (c < 128) ? Wq : (c < 256) ? Wk : (c < 384) ? Wv : We;
  int cc = c & 127;
  Wt[(size_t)c * F_IN + k] = (_Float16)W[(size_t)k * HD + cc];
}

// Load four 16x32 f16 A-fragments from an f16 LDS row (node kernel).
__device__ __forceinline__ void load_a_frags_f16(const _Float16* rowp, int hi,
                                                 v16h afrag[4]) {
#pragma unroll
  for (int kt = 0; kt < 4; ++kt) {
    int k0 = kt * 32 + hi * 16;
    h8 lo  = *(const h8*)(rowp + k0);
    h8 hi8 = *(const h8*)(rowp + k0 + 8);
    afrag[kt] = __builtin_shufflevector(lo, hi8, 0, 1, 2, 3, 4, 5, 6, 7,
                                        8, 9, 10, 11, 12, 13, 14, 15);
  }
}

// ---------------------------------------------------------------------------
// Node projections: grid.x = N/16 row tiles, 128 threads (4 waves).
// Each wave runs 6 of the 24 output col-tiles (Q:0-7, K:8-15, V:16-23).
// ---------------------------------------------------------------------------
__global__ void __launch_bounds__(128) node_proj_kernel(
    const float* __restrict__ X, const _Float16* __restrict__ Wt,
    const float* __restrict__ bq, const float* __restrict__ bk,
    const float* __restrict__ bv,
    float* __restrict__ Qh, float* __restrict__ Kh, float* __restrict__ Vh) {
  __shared__ _Float16 sA[16][136];   // +8 halfs row pad -> conflict-free b128
  const int mtile = blockIdx.x;
  const int t = threadIdx.x;

  const float* srcp = X + (size_t)mtile * 16 * F_IN;
#pragma unroll
  for (int i = 0; i < 16; ++i) {
    int lin = t + i * 128;
    sA[lin >> 7][lin & 127] = (_Float16)srcp[lin];
  }
  __syncthreads();

  const int wave = t >> 5, lane = t & 31;
  const int nl = lane & 15;
  const int hi = lane >> 4;

  v16h afrag[4];
  load_a_frags_f16(&sA[nl][0], hi, afrag);

#pragma unroll
  for (int j = 0; j < 6; ++j) {
    int c = wave + j * 4;            // col-tile 0..23
    const _Float16* wp = Wt + (size_t)(c * 16 + nl) * F_IN;
    v8f acc = {};
#pragma unroll
    for (int kt = 0; kt < 4; ++kt) {
      v16h bfrag = *(const v16h*)(wp + kt * 32 + hi * 16);
      acc = __builtin_amdgcn_wmma_f32_16x16x32_f16(
          false, afrag[kt], false, bfrag, (short)0, acc, false, false);
    }
    int mat = c >> 3;                // 0=Q 1=K 2=V
    int colbase = (c & 7) * 16;
    const float* bp = (mat == 0) ? bq : (mat == 1) ? bk : bv;
    float* outp = (mat == 0) ? Qh : (mat == 1) ? Kh : Vh;
    float bias = bp[colbase + nl];
#pragma unroll
    for (int r = 0; r < 8; ++r) {
      int m = hi ? (r + 8) : r;      // C/D layout: VGPR r -> M=r (+8 hi half)
      outp[(size_t)(mtile * 16 + m) * HD + colbase + nl] = acc[r] + bias;
    }
  }
}

// ---------------------------------------------------------------------------
// Edge kernel: grid.x = E/(16*ET_PER_BLK), 128 threads (4 waves).
// edge_feats is streamed HBM->LDS with GLOBAL_LOAD_ASYNC_TO_LDS_B128 into a
// 2-deep double buffer (ASYNCcnt-tracked), overlapping the next tile's DMA
// with this tile's WMMA + gather/scatter math. fp32 stays raw in LDS; each
// wave converts to f16 while building A-fragments.
// Each col-tile of 16 == one head (D=16). Wave w handles heads {w, w+4}.
// ---------------------------------------------------------------------------
__global__ void __launch_bounds__(128) edge_kernel(
    const float* __restrict__ EF, const _Float16* __restrict__ Wt,
    const float* __restrict__ be,
    const int* __restrict__ srcI, const int* __restrict__ dstI,
    const float* __restrict__ Qh, const float* __restrict__ Kh,
    const float* __restrict__ Vh,
    float* __restrict__ e_out, float* __restrict__ hacc,
    float* __restrict__ zacc) {
  __shared__ float sBuf[2][16][132];  // 528B row stride: 16B-aligned, padded
  const int t = threadIdx.x;
  const int wave = t >> 5, lane = t & 31;
  const int nl = lane & 15;
  const int hi = lane >> 4;
  const int tile0 = blockIdx.x * ET_PER_BLK;

  // Issue one tile's async DMA: 512 x 16B chunks, 4 per thread.
  auto issue_tile = [&](int tileIdx, int buf) {
    const char* gbase = (const char*)EF + (size_t)tileIdx * 16 * 512;
#pragma unroll
    for (int q = 0; q < 4; ++q) {
      int c = q * 128 + t;             // chunk id 0..511
      int row = c >> 5;
      int colB = (c & 31) * 16;        // byte offset within row
      unsigned lds = lds_off32(&sBuf[buf][row][0]) + (unsigned)colB;
      unsigned long long ga =
          (unsigned long long)(gbase + (size_t)row * 512 + colB);
      asm volatile("global_load_async_to_lds_b128 %0, %1, off"
                   :: "v"(lds), "v"(ga) : "memory");
    }
  };

  issue_tile(tile0 + 0, 0);
  issue_tile(tile0 + 1, 1);

  for (int i = 0; i < ET_PER_BLK; ++i) {
    const int buf = i & 1;
    const int etile = tile0 + i;

    // Wait for tile i to land (next tile's 4 async ops may stay in flight;
    // async loads complete in order).
    if (i + 1 < ET_PER_BLK)
      asm volatile("s_wait_asynccnt 0x4" ::: "memory");
    else
      asm volatile("s_wait_asynccnt 0x0" ::: "memory");
    __syncthreads();

    // Build A-fragments (fp32 LDS -> f16) for this tile.
    const float* rowp = &sBuf[buf][nl][0];
    v16h afrag[4];
#pragma unroll
    for (int kt = 0; kt < 4; ++kt) {
      int k0 = kt * 32 + hi * 16;
      f4 a0 = *(const f4*)(rowp + k0);
      f4 a1 = *(const f4*)(rowp + k0 + 4);
      f4 a2 = *(const f4*)(rowp + k0 + 8);
      f4 a3 = *(const f4*)(rowp + k0 + 12);
      v16h fr;
#pragma unroll
      for (int x = 0; x < 4; ++x) {
        fr[x]      = (_Float16)a0[x];
        fr[4 + x]  = (_Float16)a1[x];
        fr[8 + x]  = (_Float16)a2[x];
        fr[12 + x] = (_Float16)a3[x];
      }
      afrag[kt] = fr;
    }

#pragma unroll
    for (int j = 0; j < 2; ++j) {
      int h = wave + j * 4;            // head 0..7
      const _Float16* wp = Wt + (size_t)(384 + h * 16 + nl) * F_IN;
      v8f acc = {};
#pragma unroll
      for (int kt = 0; kt < 4; ++kt) {
        v16h bfrag = *(const v16h*)(wp + kt * 32 + hi * 16);
        acc = __builtin_amdgcn_wmma_f32_16x16x32_f16(
            false, afrag[kt], false, bfrag, (short)0, acc, false, false);
      }
      float bias = be[h * 16 + nl];
#pragma unroll
      for (int r = 0; r < 8; ++r) {
        int m = hi ? (r + 8) : r;
        int e = etile * 16 + m;
        int s  = srcI[e];               // L2-resident (6.4MB index arrays)
        int d2 = dstI[e];
        size_t col = (size_t)h * 16 + nl;
        float proj = acc[r] + bias;
        float kv = Kh[(size_t)s  * HD + col];   // L2-resident gathers
        float qv = Qh[(size_t)d2 * HD + col];
        float sc = fminf(fmaxf(kv * qv * 0.25f, -5.0f), 5.0f) * proj;
        e_out[(size_t)e * HD + col] = sc;
        // reduce over D=16 lanes within half-wave (xor stays in 16-group)
        float sum = sc;
        sum += __shfl_xor(sum, 1, 32);
        sum += __shfl_xor(sum, 2, 32);
        sum += __shfl_xor(sum, 4, 32);
        sum += __shfl_xor(sum, 8, 32);
        float w = __expf(fminf(fmaxf(sum, -5.0f), 5.0f));
        float vv = Vh[(size_t)s * HD + col];
        atomicAdd(&hacc[(size_t)d2 * HD + col], w * vv);  // segsum (L2)
        if (nl == 0) atomicAdd(&zacc[(size_t)d2 * 8 + h], w);
      }
    }

    __syncthreads();                    // all waves done reading sBuf[buf]
    if (i + 2 < ET_PER_BLK) issue_tile(tile0 + i + 2, buf);
  }
}

// ---------------------------------------------------------------------------
// Finalize: h_out = wV / (z + 1e-8), in place on d_out's h region.
// ---------------------------------------------------------------------------
__global__ void __launch_bounds__(256) finalize_kernel(
    float* __restrict__ hacc, const float* __restrict__ zacc, int total) {
  int i = blockIdx.x * blockDim.x + threadIdx.x;
  if (i < total) {
    int n = i >> 7;
    int h = (i >> 4) & 7;
    hacc[i] = hacc[i] / (zacc[n * 8 + h] + 1e-8f);
  }
}

extern "C" void kernel_launch(void* const* d_in, const int* in_sizes, int n_in,
                              void* d_out, int out_size, void* d_ws,
                              size_t ws_size, hipStream_t stream) {
  const float* node = (const float*)d_in[0];
  const float* edge = (const float*)d_in[1];
  const int* srcI = (const int*)d_in[2];
  const int* dstI = (const int*)d_in[3];
  const float* Wq = (const float*)d_in[4];
  const float* bq = (const float*)d_in[5];
  const float* Wk = (const float*)d_in[6];
  const float* bk = (const float*)d_in[7];
  const float* Wv = (const float*)d_in[8];
  const float* bv = (const float*)d_in[9];
  const float* We = (const float*)d_in[10];
  const float* be = (const float*)d_in[11];

  float* out  = (float*)d_out;
  float* hout = out;                               // [N, 128] (wV accum)
  float* eout = out + (size_t)N_NODES * HD;        // [E, 128]

  char* ws = (char*)d_ws;
  _Float16* Wt = (_Float16*)ws;                    // 512*128 f16 = 128 KB
  size_t off = 512 * F_IN * sizeof(_Float16);
  float* Qh = (float*)(ws + off);  off += (size_t)N_NODES * HD * sizeof(float);
  float* Kh = (float*)(ws + off);  off += (size_t)N_NODES * HD * sizeof(float);
  float* Vh = (float*)(ws + off);  off += (size_t)N_NODES * HD * sizeof(float);
  float* zacc = (float*)(ws + off);                // [N, 8]

  // zero the segment-sum accumulators (graph-capture-safe)
  hipMemsetAsync(hout, 0, (size_t)N_NODES * HD * sizeof(float), stream);
  hipMemsetAsync(zacc, 0, (size_t)N_NODES * 8 * sizeof(float), stream);

  prep_weights_kernel<<<512, 128, 0, stream>>>(Wq, Wk, Wv, We, Wt);
  node_proj_kernel<<<N_NODES / 16, 128, 0, stream>>>(node, Wt, bq, bk, bv,
                                                     Qh, Kh, Vh);
  edge_kernel<<<N_EDGES / (16 * ET_PER_BLK), 128, 0, stream>>>(
      edge, Wt, be, srcI, dstI, Qh, Kh, Vh, eout, hout, zacc);
  int total = N_NODES * HD;
  finalize_kernel<<<(total + 255) / 256, 256, 0, stream>>>(hout, zacc, total);
}
